// ADesign_61804579389537
// MI455X (gfx1250) — matrix operand order repaired and checked by reading, not compiled, not run on hardware
//
#include <hip/hip_runtime.h>
#include <math.h>

#define H 128
#define NN 5000
#define NE 150000
#define VOCAB 20

typedef __attribute__((ext_vector_type(16))) __bf16 v16bf;
typedef __attribute__((ext_vector_type(8)))  float  v8f;
typedef __attribute__((ext_vector_type(4)))  float  fx4;   // native vec4 (AS-agnostic)

#define AS1 __attribute__((address_space(1)))
typedef const AS1 float*          gfp;    // global fp32 pointer
typedef const AS1 fx4*            gf4p;   // global vec4 pointer
typedef const AS1 unsigned short* gusp;   // global bf16-bits pointer

struct U16x16 { unsigned short u[16]; };   // 32B, bit_cast target for v16bf
struct F4x2   { fx4 a; fx4 b; };           // 32B

struct GSeg  { const float* src; const int* idx; int cols; int shift; int rows; };
struct GSpec { GSeg s[5]; int nseg; int K; };

__device__ __forceinline__ unsigned short f2bf(float f){
  union { float f; unsigned u; } x; x.f = f;
  unsigned r = x.u + 0x7FFFu + ((x.u >> 16) & 1u);   // round-to-nearest-even
  return (unsigned short)(r >> 16);
}
// hot-path conversion: native bf16 cvt (RTNE) -> v_cvt*_bf16_f32
__device__ __forceinline__ v16bf pack16(const float* f){
  v16bf r;
#pragma unroll
  for (int j = 0; j < 16; ++j) r[j] = (__bf16)f[j];
  return r;
}

// ---------------- weight conversion into B-fragment-swizzled bf16 ----------------
// element o = ((nt*nch + ck)*2 + h)*256 + nl*16 + kk  <->  W[k=ck*32+h*16+kk][n=nt*16+nl]
__global__ void k_w2bfsw(const float* __restrict__ w, unsigned short* __restrict__ o,
                         int K, int N, int Npad, int nch){
  int i = blockIdx.x*blockDim.x + threadIdx.x;
  int total = Npad*nch*32;
  if (i >= total) return;
  int kk = i & 15, nl = (i >> 4) & 15, h = (i >> 8) & 1, t2 = i >> 9;
  int ck = t2 % nch, nt = t2 / nch;
  int k = ck*32 + h*16 + kk;
  int n = nt*16 + nl;
  float v = (k < K && n < N) ? w[(long)k*N + n] : 0.f;
  o[i] = f2bf(v);
}
// conv weight [O][I][T] as GEMM weight W[k=t*I+ic][n=o], swizzled as above (N=Npad=O)
__global__ void k_convw2bfsw(const float* __restrict__ w, unsigned short* __restrict__ o,
                             int O, int I, int T, int nch){
  int i = blockIdx.x*blockDim.x + threadIdx.x;
  int total = O*nch*32;
  if (i >= total) return;
  int kk = i & 15, nl = (i >> 4) & 15, h = (i >> 8) & 1, t2 = i >> 9;
  int ck = t2 % nch, nt = t2 / nch;
  int k = ck*32 + h*16 + kk;
  int n = nt*16 + nl;
  float v = 0.f;
  if (k < I*T && n < O){
    int t = k / I, ic = k % I;
    v = w[((long)n*I + ic)*T + t];
  }
  o[i] = f2bf(v);
}

// ---------------- LDS-free streaming WMMA GEMM with gathered A ----------------
// Out[M,Nvalid] = postscale * act( gatherA[M,K] @ W + bias + resid )
// 256 threads = 8 waves, each wave owns 16 rows x NT*16 cols; K chunked by 32.
// All fragment traffic via explicit addrspace(1) pointers -> global_load_b128.
template<int NT>
__global__ __launch_bounds__(256)
void k_gemm(GSpec gs, const unsigned short* __restrict__ W, int nch,
            const float* __restrict__ bias, const float* __restrict__ resid,
            float* __restrict__ out, int M, int Nvalid, int act, float postscale)
{
  const int lane = threadIdx.x & 31;
  const int wvid = threadIdx.x >> 5;
  const int nl   = lane & 15;
  const int hk   = (lane < 16) ? 0 : 16;   // A: K 0..15 vs 16..31
  const long hB  = (lane < 16) ? 0 : 1;    // B: K 0..15 vs 16..31
  const int nbase = blockIdx.y * (NT*16);
  const int nt0   = nbase >> 4;
  const int mrow  = blockIdx.x*128 + wvid*16 + nl;
  v8f acc[NT] = {};

  // per-ntile weight pointers (global AS); +512 elements per 32-K chunk
  gusp bptr[NT];
#pragma unroll
  for (int nt = 0; nt < NT; ++nt)
    bptr[nt] = (gusp)(unsigned long long)
               (W + (((long)(nt0 + nt)*nch*2 + hB) << 8) + (nl << 4));

  // incremental segment state (k0 monotonically increasing, chunks never straddle)
  int sidx = -1, segbase = 0, segend = 0, segcols = 0;
  bool fastseg = false;
  gfp rowbase = nullptr;   // gathered row pointer for current segment (global AS)

  for (int ck = 0; ck < nch; ++ck){
    const int k0 = ck*32;
    if (k0 >= segend){
      do { segbase = segend; ++sidx; segend += gs.s[sidx].cols; } while (k0 >= segend);
      const GSeg sg = gs.s[sidx];
      segcols = sg.cols;
      fastseg = ((segcols & 3) == 0);
      rowbase = nullptr;
      if (mrow < M){
        int r = sg.idx ? sg.idx[mrow] : (mrow + sg.shift);
        if (r >= 0 && r < sg.rows)
          rowbase = (gfp)(unsigned long long)(sg.src) + (long)r*segcols;
      }
    }
    const int scol0 = k0 - segbase;
    gfp rowp = rowbase ? (rowbase + scol0) : (gfp) nullptr;
    // ---- A fragment: direct global gather + register bf16 convert ----
    v16bf afrag;
    if (rowp && fastseg && (scol0 + 32 <= segcols)){
      fx4 f0 = ((gf4p)(rowp + hk))[0];        // K hk..hk+3
      fx4 f1 = ((gf4p)(rowp + hk))[1];        // K hk+4..hk+7
      fx4 f2 = ((gf4p)(rowp + 8 + hk))[0];   // K hk+8..
      fx4 f3 = ((gf4p)(rowp + 8 + hk))[1];
      float f[16] = { f0[0],f0[1],f0[2],f0[3], f1[0],f1[1],f1[2],f1[3],
                      f2[0],f2[1],f2[2],f2[3], f3[0],f3[1],f3[2],f3[3] };
      afrag = pack16(f);
    } else if (rowp){
      float f[16];
#pragma unroll
      for (int j = 0; j < 8; ++j){
        int ka = hk + j, kb = 8 + hk + j;
        f[j]   = (scol0 + ka < segcols) ? rowp[ka] : 0.f;
        f[8+j] = (scol0 + kb < segcols) ? rowp[kb] : 0.f;
      }
      afrag = pack16(f);
    } else {
      U16x16 z{};
      afrag = __builtin_bit_cast(v16bf, z);
    }
    // ---- B fragments straight from swizzled weights (coalesced global b128) ----
    v16bf bfr[NT];
#pragma unroll
    for (int nt = 0; nt < NT; ++nt){
      gusp bp = bptr[nt];
      F4x2 t; t.a = ((gf4p)bp)[0]; t.b = ((gf4p)(bp + 8))[0];
      bfr[nt] = __builtin_bit_cast(v16bf, t);
      bptr[nt] = bp + 512;
    }
    // ---- branch-free WMMA chain ----
#pragma unroll
    for (int nt = 0; nt < NT; ++nt)
      acc[nt] = __builtin_amdgcn_wmma_f32_16x16x32_bf16(
          false, afrag, false, bfr[nt], (short)0, acc[nt], false, false);
  }
  // ---- epilogue: C layout vgpr r -> M = r + (lane<16?0:8), N = lane%16 ----
  const int rowoff = (lane < 16) ? 0 : 8;
#pragma unroll
  for (int nt = 0; nt < NT; ++nt){
    int c = nbase + nt*16 + nl;
    if (c >= Nvalid) continue;
#pragma unroll
    for (int r = 0; r < 8; ++r){
      int m = blockIdx.x*128 + wvid*16 + rowoff + r;
      if (m >= M) continue;
      float v = acc[nt][r];
      if (bias)  v += bias[c];
      if (resid) v += resid[(long)m*Nvalid + c];
      if (act == 1) v = fmaxf(v, 0.f);
      else if (act == 2) v = (v > 0.f) ? v : 0.01f*v;
      v *= postscale;
      out[(long)m*Nvalid + c] = v;
    }
  }
}

// ---------------- BatchNorm (training-mode batch stats), C == 128 ----------------
__global__ void k_bn_stats(const float* __restrict__ x, float* __restrict__ st, long M){
  long total = M*128;
  long stride = (long)gridDim.x * blockDim.x;   // multiple of 128
  float s = 0.f, s2 = 0.f;
  for (long i = (long)blockIdx.x*blockDim.x + threadIdx.x; i < total; i += stride){
    float v = x[i]; s += v; s2 += v*v;
  }
  __shared__ float sh[512];
  int t = threadIdx.x;
  sh[t] = s; sh[256 + t] = s2;
  __syncthreads();
  if (t < 128){
    atomicAdd(&st[t],       sh[t] + sh[t+128]);
    atomicAdd(&st[128 + t], sh[256+t] + sh[256+t+128]);
  }
}
__global__ void k_bn_fin(float* st, float invM){
  int c = threadIdx.x;
  float mu  = st[c] * invM;
  float var = st[128+c] * invM - mu*mu;
  st[c] = mu; st[128+c] = rsqrtf(var + 1e-5f);
}
__global__ void k_bn_apply(float* x, const float* __restrict__ g, const float* __restrict__ b,
                           const float* __restrict__ st, long M, int relu){
  long i = (long)blockIdx.x*blockDim.x + threadIdx.x;
  if (i >= M*128) return;
  int c = (int)(i & 127);
  float v = g[c]*(x[i] - st[c])*st[128+c] + b[c];
  if (relu) v = fmaxf(v, 0.f);
  x[i] = v;
}

// ---------------- misc elementwise ----------------
__global__ void k_fill(float* p, float v, long n){
  long i = (long)blockIdx.x*blockDim.x + threadIdx.x;
  if (i < n) p[i] = v;
}
__global__ void k_add(float* a, const float* b, long n){
  long i = (long)blockIdx.x*blockDim.x + threadIdx.x;
  if (i < n) a[i] += b[i];
}

// ---------------- segment softmax attention ----------------
__device__ __forceinline__ void atomicMaxF(float* addr, float val){
  int* ia = (int*)addr;
  int cur = __float_as_int(*addr);
  while (__int_as_float(cur) < val){
    int old = atomicCAS(ia, cur, __float_as_int(val));
    if (old == cur) break;
    cur = old;
  }
}
__global__ void k_att_max(const float* __restrict__ w3, const int* __restrict__ center,
                          float* m, int E){
  int i = blockIdx.x*blockDim.x + threadIdx.x;
  if (i >= E*4) return;
  int e = i >> 2, h = i & 3;
  atomicMaxF(&m[center[e]*4 + h], w3[i]);
}
__global__ void k_att_expsum(const float* __restrict__ w3, const int* __restrict__ center,
                             const float* __restrict__ m, float* s, float* att, int E){
  int i = blockIdx.x*blockDim.x + threadIdx.x;
  if (i >= E*4) return;
  int e = i >> 2, h = i & 3;
  int c = center[e]*4 + h;
  float v = expf(w3[i] - m[c]);
  att[i] = v;
  atomicAdd(&s[c], v);
}
__global__ void k_att_agg(const float* __restrict__ att, const float* __restrict__ s,
                          const int* __restrict__ center, const float* __restrict__ V,
                          float* agg, int E){
  long i = (long)blockIdx.x*blockDim.x + threadIdx.x;
  if (i >= (long)E*128) return;
  int e = (int)(i >> 7), c = (int)(i & 127), h = c >> 5;
  int node = center[e];
  float a = att[e*4 + h] / s[node*4 + h];
  atomicAdd(&agg[(long)node*128 + c], a * V[i]);
}

// ---------------- decoder tail ----------------
__global__ void k_lsm(const float* __restrict__ logits, float* __restrict__ out, int M){
  int r = blockIdx.x*blockDim.x + threadIdx.x;
  if (r >= M) return;
  const float* p = logits + (long)r*VOCAB;
  float mx = p[0];
  for (int i = 1; i < VOCAB; ++i) mx = fmaxf(mx, p[i]);
  float s = 0.f;
  for (int i = 0; i < VOCAB; ++i) s += expf(p[i] - mx);
  float ls = logf(s);
  for (int i = 0; i < VOCAB; ++i) out[(long)r*VOCAB + i] = p[i] - mx - ls;
}
__global__ void k_conf(const float* __restrict__ logits, int* __restrict__ conf, int M){
  int r = blockIdx.x*blockDim.x + threadIdx.x;
  if (r >= M) return;
  const float* p = logits + (long)r*VOCAB;
  float t0 = -3.0e38f, t1 = -3.0e38f;
  for (int i = 0; i < VOCAB; ++i){
    float v = p[i];
    if (v > t0){ t1 = t0; t0 = v; } else if (v > t1){ t1 = v; }
  }
  int c = (int)(t0 / (t1 + 1e-5f));   // trunc toward zero, like astype(int32)
  if (c < 0) c = 0; if (c > 49) c = 49;
  conf[r] = c;
}
__global__ void k_xcat(const float* __restrict__ hv, const float* __restrict__ cemb,
                       const int* __restrict__ conf, float* __restrict__ xcat, int M){
  long i = (long)blockIdx.x*blockDim.x + threadIdx.x;
  if (i >= (long)M*256) return;
  int r = (int)(i >> 8), c = (int)(i & 255);
  xcat[i] = (c < 128) ? hv[(long)r*128 + c]
                      : cemb[(long)conf[r]*128 + (c - 128)];
}

// =====================================================================
extern "C" void kernel_launch(void* const* d_in, const int* in_sizes, int n_in,
                              void* d_out, int out_size, void* d_ws, size_t ws_size,
                              hipStream_t stream)
{
  (void)in_sizes; (void)n_in; (void)out_size; (void)ws_size;
  const float* h_V   = (const float*)d_in[0];
  const float* h_P   = (const float*)d_in[1];
  const int*   P_idx = (const int*)d_in[2];
  const int* center  = P_idx;
  const int* srcIdx  = P_idx + NE;
  auto F = [&](int i)->const float* { return (const float*)d_in[i]; };
  // jax pytree flatten order (dict keys sorted):
  // 4 We_b,5 We_w,6..9 Wv_bn{1b,1g,2b,2g},10..15 Wv_l{1,1b,2,2b,3,3b},
  // 16..27 dec{bn1b,bn1g,bn2b,bn2g,c1,c1b,c2,c2b,c3,c3b,ro,rob},
  // 28..40 dec2{...,conf_emb,ro,rob}, 41 edge_emb_b,42 edge_emb_w,
  // 43..138 enc[6]{B1,B1b,B2,B2b,B3,B3b,WO,WV,d1,d1b,d2,d2b,n1b,n1g,n2b,n2g},
  // 139 node_emb_b,140 node_emb_w,141/142 norm_edges_{b,g},143/144 norm_nodes_{b,g}

  char* wsb = (char*)d_ws; size_t off = 0;
  auto alloc = [&](size_t bytes)->void* {
    void* p = wsb + off; off = (off + bytes + 255) & ~(size_t)255; return p; };

  auto cvt = [&](const float* w, int K, int N, int Npad)->const unsigned short* {
    int nch = (K + 31)/32;
    size_t total = (size_t)Npad*nch*32;
    unsigned short* o = (unsigned short*)alloc(total*2);
    k_w2bfsw<<<(unsigned)((total+255)/256), 256, 0, stream>>>(w, o, K, N, Npad, nch);
    return o; };
  auto cvtc = [&](const float* w, int O, int I, int T)->const unsigned short* {
    int nch = (I*T + 31)/32;
    size_t total = (size_t)O*nch*32;
    unsigned short* o = (unsigned short*)alloc(total*2);
    k_convw2bfsw<<<(unsigned)((total+255)/256), 256, 0, stream>>>(w, o, O, I, T, nch);
    return o; };

  // ---- weights -> swizzled bf16 ----
  const unsigned short* Wnode = cvt(F(140), 12, 128, 128);
  const unsigned short* Wl1 = cvt(F(10),128,128,128);
  const unsigned short* Wl2 = cvt(F(12),128,128,128);
  const unsigned short* Wl3 = cvt(F(14),128,128,128);
  const unsigned short* Wee = cvt(F(42), 23,128,128);
  const unsigned short* WWe = cvt(F(5), 128,128,128);
  struct Mod { const unsigned short *B1,*B2,*B3,*WV,*WO,*d1,*d2;
               const float *B1b,*B2b,*B3b,*d1b,*d2b,*n1b,*n1g,*n2b,*n2g; };
  Mod mod[6];
  for (int m = 0; m < 6; ++m){
    int b = 43 + m*16;
    mod[m].B1  = cvt(F(b+0), 384,128,128);  mod[m].B1b = F(b+1);
    mod[m].B2  = cvt(F(b+2), 128,128,128);  mod[m].B2b = F(b+3);
    mod[m].B3  = cvt(F(b+4), 128,  4, 16);  mod[m].B3b = F(b+5);
    mod[m].WO  = cvt(F(b+6), 128,128,128);
    mod[m].WV  = cvt(F(b+7), 256,128,128);
    mod[m].d1  = cvt(F(b+8), 128,512,512);  mod[m].d1b = F(b+9);
    mod[m].d2  = cvt(F(b+10),512,128,128);  mod[m].d2b = F(b+11);
    mod[m].n1b = F(b+12); mod[m].n1g = F(b+13);
    mod[m].n2b = F(b+14); mod[m].n2g = F(b+15);
  }
  const unsigned short* D1c1 = cvtc(F(20),128,128,5);
  const unsigned short* D1c2 = cvtc(F(22),128,128,5);
  const unsigned short* D1c3 = cvtc(F(24),128,128,5);
  const unsigned short* D1ro = cvt (F(26),128, 20, 32);
  const unsigned short* D2c1 = cvtc(F(32),128,256,5);
  const unsigned short* D2c2 = cvtc(F(34),128,128,5);
  const unsigned short* D2c3 = cvtc(F(36),128,128,5);
  const unsigned short* D2ro = cvt (F(39),128, 20, 32);

  // ---- activations ----
  float* hvA   = (float*)alloc((size_t)NN*128*4);
  float* hvB   = (float*)alloc((size_t)NN*128*4);
  float* ntmp  = (float*)alloc((size_t)NN*128*4);
  float* t512  = (float*)alloc((size_t)NN*512*4);
  float* agg   = (float*)alloc((size_t)NN*128*4);
  float* hp    = (float*)alloc((size_t)NE*128*4);
  float* E1    = (float*)alloc((size_t)NE*128*4);
  float* E2    = (float*)alloc((size_t)NE*128*4);
  float* w3    = (float*)alloc((size_t)NE*4*4);
  float* attw  = (float*)alloc((size_t)NE*4*4);
  float* attm  = (float*)alloc((size_t)NN*4*4);
  float* atts  = (float*)alloc((size_t)NN*4*4);
  float* bnstat= (float*)alloc(256*4);
  float* logitsb=(float*)alloc((size_t)NN*VOCAB*4);
  int*   confb = (int*)  alloc((size_t)NN*4);
  float* xcat  = (float*)alloc((size_t)NN*256*4);
  float* db1   = (float*)alloc((size_t)NN*128*4);
  float* db2   = (float*)alloc((size_t)NN*128*4);

  auto gemm = [&](const GSpec& gs, const unsigned short* W, const float* bias,
                  const float* resid, float* out, int M, int Npad, int Nvalid,
                  int act, float ps){
    int nch = (gs.K + 31)/32;
    if (Npad >= 128){
      dim3 g((M+127)/128, Npad/128);
      k_gemm<8><<<g, 256, 0, stream>>>(gs, W, nch, bias, resid, out, M, Nvalid, act, ps);
    } else if (Npad == 32){
      dim3 g((M+127)/128, 1);
      k_gemm<2><<<g, 256, 0, stream>>>(gs, W, nch, bias, resid, out, M, Nvalid, act, ps);
    } else {
      dim3 g((M+127)/128, 1);
      k_gemm<1><<<g, 256, 0, stream>>>(gs, W, nch, bias, resid, out, M, Nvalid, act, ps);
    }
  };
  auto bn = [&](float* x, const float* g, const float* b, long M, int relu){
    (void)hipMemsetAsync(bnstat, 0, 256*4, stream);
    k_bn_stats<<<256, 256, 0, stream>>>(x, bnstat, M);
    k_bn_fin<<<1, 128, 0, stream>>>(bnstat, 1.0f/(float)M);
    long tot = M*128;
    k_bn_apply<<<(unsigned)((tot+255)/256), 256, 0, stream>>>(x, g, b, bnstat, M, relu);
  };
  auto fill = [&](float* p, float v, long n){
    k_fill<<<(unsigned)((n+255)/256), 256, 0, stream>>>(p, v, n); };
  auto S1 = [](const float* p, int cols, int rows)->GSpec{
    GSpec g{}; g.s[0] = GSeg{p, nullptr, cols, 0, rows}; g.nseg = 1; g.K = cols; return g; };
  auto Sconv = [](const float* p, int cols, int rows)->GSpec{
    GSpec g{};
    for (int t = 0; t < 5; ++t) g.s[t] = GSeg{p, nullptr, cols, t-2, rows};
    g.nseg = 5; g.K = cols*5; return g; };

  const float rs32 = 0.17677669529663687f;  // 1/sqrt(H/HEADS)
  auto gnn = [&](const Mod& md, const float* hv_in, float* hv_out){
    // x = [hv[center] | hp | hv[src]] (gathered, never materialized)
    GSpec gB1{};
    gB1.s[0] = GSeg{hv_in, center, 128, 0, NN};
    gB1.s[1] = GSeg{hp,   nullptr, 128, 0, NE};
    gB1.s[2] = GSeg{hv_in, srcIdx, 128, 0, NN};
    gB1.nseg = 3; gB1.K = 384;
    gemm(gB1, md.B1, md.B1b, nullptr, E1, NE, 128, 128, 1, 1.0f);
    gemm(S1(E1,128,NE), md.B2, md.B2b, nullptr, E2, NE, 128, 128, 1, 1.0f);
    gemm(S1(E2,128,NE), md.B3, md.B3b, nullptr, w3, NE,  16,   4, 0, rs32);
    GSpec gWV{};
    gWV.s[0] = GSeg{hp,   nullptr, 128, 0, NE};
    gWV.s[1] = GSeg{hv_in, srcIdx, 128, 0, NN};
    gWV.nseg = 2; gWV.K = 256;
    gemm(gWV, md.WV, nullptr, nullptr, E1, NE, 128, 128, 0, 1.0f);   // V
    fill(attm, -3.0e38f, (long)NN*4);
    fill(atts, 0.f, (long)NN*4);
    fill(agg,  0.f, (long)NN*128);
    k_att_max   <<<(NE*4+255)/256, 256, 0, stream>>>(w3, center, attm, NE);
    k_att_expsum<<<(NE*4+255)/256, 256, 0, stream>>>(w3, center, attm, atts, attw, NE);
    k_att_agg<<<(unsigned)(((long)NE*128+255)/256), 256, 0, stream>>>(attw, atts, center, E1, agg, NE);
    gemm(S1(agg,128,NN), md.WO, nullptr, hv_in, ntmp, NN, 128, 128, 0, 1.0f);
    bn(ntmp, md.n1g, md.n1b, NN, 0);
    gemm(S1(ntmp,128,NN), md.d1, md.d1b, nullptr, t512,  NN, 512, 512, 1, 1.0f);
    gemm(S1(t512,512,NN), md.d2, md.d2b, ntmp,   hv_out, NN, 128, 128, 0, 1.0f);
    bn(hv_out, md.n2g, md.n2b, NN, 0);
  };

  // ---- node embedding + MLP ----
  gemm(S1(h_V,12,NN), Wnode, F(139), nullptr, hvA, NN,128,128, 0, 1.f);
  bn(hvA, F(144), F(143), NN, 0);
  gemm(S1(hvA,128,NN), Wl1, F(11), nullptr, hvB, NN,128,128, 2, 1.f);
  bn(hvB, F(7), F(6), NN, 0);
  gemm(S1(hvB,128,NN), Wl2, F(13), nullptr, hvA, NN,128,128, 2, 1.f);
  bn(hvA, F(9), F(8), NN, 0);
  gemm(S1(hvA,128,NN), Wl3, F(15), nullptr, hvB, NN,128,128, 0, 1.f);
  float* hv = hvB; float* hvo = hvA;
  // ---- edge embedding ----
  gemm(S1(h_P,23,NE), Wee, F(41), nullptr, E1, NE,128,128, 0, 1.f);
  bn(E1, F(142), F(141), NE, 0);
  gemm(S1(E1,128,NE), WWe, F(4), nullptr, hp, NE,128,128, 0, 1.f);
  // ---- 3 encoder layers, 2 GNN modules each (2nd with outer residual) ----
  for (int L = 0; L < 3; ++L){
    gnn(mod[L*2+0], hv, hvo);
    { float* t = hv; hv = hvo; hvo = t; }
    gnn(mod[L*2+1], hv, hvo);
    k_add<<<(unsigned)(((long)NN*128+255)/256), 256, 0, stream>>>(hv, hvo, (long)NN*128);
  }
  float* lp  = (float*)d_out;                     // dec2 output (first in tuple)
  float* lp0 = (float*)d_out + (size_t)NN*VOCAB;  // dec output
  // ---- decoder 1 (conv1d k=5 as 5-shift gather GEMM) ----
  gemm(Sconv(hv,128,NN),  D1c1, F(21), nullptr, db1, NN,128,128, 0, 1.f);
  bn(db1, F(17), F(16), NN, 1);
  gemm(Sconv(db1,128,NN), D1c2, F(23), nullptr, db2, NN,128,128, 0, 1.f);
  bn(db2, F(19), F(18), NN, 1);
  gemm(Sconv(db2,128,NN), D1c3, F(25), nullptr, db1, NN,128,128, 0, 1.f);
  gemm(S1(db1,128,NN),    D1ro, F(27), nullptr, logitsb, NN,32,VOCAB, 0, 1.f);
  k_lsm <<<(NN+255)/256, 256, 0, stream>>>(logitsb, lp0, NN);
  k_conf<<<(NN+255)/256, 256, 0, stream>>>(logitsb, confb, NN);
  k_xcat<<<(unsigned)(((long)NN*256+255)/256), 256, 0, stream>>>(hv, F(38), confb, xcat, NN);
  // ---- decoder 2 ----
  gemm(Sconv(xcat,256,NN), D2c1, F(33), nullptr, db1, NN,128,128, 0, 1.f);
  bn(db1, F(29), F(28), NN, 1);
  gemm(Sconv(db1,128,NN),  D2c2, F(35), nullptr, db2, NN,128,128, 0, 1.f);
  bn(db2, F(31), F(30), NN, 1);
  gemm(Sconv(db2,128,NN),  D2c3, F(37), nullptr, db1, NN,128,128, 0, 1.f);
  gemm(S1(db1,128,NN),     D2ro, F(40), nullptr, logitsb, NN,32,VOCAB, 0, 1.f);
  k_lsm<<<(NN+255)/256, 256, 0, stream>>>(logitsb, lp, NN);
}
